// PhyHGkNN4_6897717477693
// MI455X (gfx1250) — matrix operand order repaired and checked by reading, not compiled, not run on hardware
//
#include <hip/hip_runtime.h>
#include <cmath>

constexpr int B_  = 8;
constexpr int N_  = 8192;
constexpr int K1_ = 64;
constexpr int K2_ = 128;
constexpr int M_  = 256;   // 2*K1 + K2
constexpr int C_  = 128;
constexpr int KM_ = 16;
constexpr int L_  = 3;

typedef __attribute__((ext_vector_type(16))) __bf16 v16bf;
typedef __attribute__((ext_vector_type(8)))  float  v8f;
typedef __attribute__((ext_vector_type(8)))  __bf16 bf8v;
typedef __attribute__((ext_vector_type(4)))  __bf16 bf4v;
typedef __attribute__((ext_vector_type(2)))  __bf16 bf2v;

#define WMMA_BF16(a_, b_, c_) \
  __builtin_amdgcn_wmma_f32_16x16x32_bf16(false, (a_), false, (b_), (short)0, (c_), false, false)

// ---------------------------------------------------------------------------
// Vectorized WMMA fragment loads from LDS (layouts per CDNA5 ISA 7.12.2).
// With m-major A tiles [m][k] and n-major B tiles [n][k], each lane's
// fragment is two contiguous 8 x bf16 runs -> 2x ds_load_b128 per fragment.
//   A 16x32: lane m=lane&15, half=lane>>4; K in [half*8, +8) and [16+half*8, +8)
//   B 32x16: lane n=lane&15, half=lane>>4; K in [half*8, +8) and [16+half*8, +8)
// ldk (elements) must be a multiple of 8 for 16B alignment.
// ---------------------------------------------------------------------------
__device__ __forceinline__ v16bf frag16(bf8v lo, bf8v hi) {
  return __builtin_shufflevector(lo, hi, 0, 1, 2, 3, 4, 5, 6, 7,
                                 8, 9, 10, 11, 12, 13, 14, 15);
}
__device__ __forceinline__ v16bf ld_a_mk2(const __bf16* p, int ldk, int m, int half) {
  bf8v lo = *(const bf8v*)&p[m * ldk + half * 8];
  bf8v hi = *(const bf8v*)&p[m * ldk + 16 + half * 8];
  return frag16(lo, hi);
}
__device__ __forceinline__ v16bf ld_b_nk2(const __bf16* p, int ldk, int n, int half) {
  bf8v lo = *(const bf8v*)&p[n * ldk + half * 8];
  bf8v hi = *(const bf8v*)&p[n * ldk + 16 + half * 8];
  return frag16(lo, hi);
}

// ---------------------------------------------------------------------------
// Bases (fast transcendental intrinsics -> single v_cos/v_sin/v_exp TRANS ops)
// ---------------------------------------------------------------------------
__device__ __forceinline__ float base_val(int k, float g0, float g1,
                                          const float* bwf, const float* pts,
                                          const float* bwg) {
  if (k < 2 * K1_) {
    int q = (k < K1_) ? k : k - K1_;
    float ph = g0 * bwf[2 * q] + g1 * bwf[2 * q + 1];
    return (k < K1_) ? __cosf(ph) : __sinf(ph);
  } else {
    int q = k - 2 * K1_;
    float w0 = fabsf(bwg[2 * q]), w1 = fabsf(bwg[2 * q + 1]);
    float d0 = g0 - pts[2 * q], d1 = g1 - pts[2 * q + 1];
    return sqrtf(w0 * w1) * __expf(-(w0 * d0 * d0 + w1 * d1 * d1));
  }
}

// 2048 blocks x 256 thr; block = 32 rows of one batch; accumulate sum of squares.
__global__ __launch_bounds__(256) void k_bases_ss(const float* __restrict__ x,
                                                  const float* __restrict__ bwf,
                                                  const float* __restrict__ pts,
                                                  const float* __restrict__ bwg,
                                                  float* __restrict__ gss) {
  __shared__ float ss[M_];
  int t = threadIdx.x;
  ss[t] = 0.f;
  __syncthreads();
  int blk = blockIdx.x;
  int b = blk >> 8;
  int n = ((blk & 255) << 5) + (t >> 3);
  int sub = t & 7;
  float g0 = x[((size_t)b * N_ + n) * 3 + 1];
  float g1 = x[((size_t)b * N_ + n) * 3 + 2];
  for (int kk = 0; kk < 32; ++kk) {
    int k = sub * 32 + kk;
    float v = base_val(k, g0, g1, bwf, pts, bwg);
    atomicAdd(&ss[k], v * v);
  }
  __syncthreads();
  atomicAdd(&gss[b * M_ + t], ss[t]);
}

// Same decomposition; recompute, scale, write bf16.
__global__ __launch_bounds__(256) void k_bases_write(const float* __restrict__ x,
                                                     const float* __restrict__ bwf,
                                                     const float* __restrict__ pts,
                                                     const float* __restrict__ bwg,
                                                     const float* __restrict__ gss,
                                                     __bf16* __restrict__ basesbf) {
  int t = threadIdx.x;
  int blk = blockIdx.x;
  int b = blk >> 8;
  int n = ((blk & 255) << 5) + (t >> 3);
  int sub = t & 7;
  float g0 = x[((size_t)b * N_ + n) * 3 + 1];
  float g1 = x[((size_t)b * N_ + n) * 3 + 2];
  __bf16 buf[32];
  float sqn = sqrtf((float)N_);
  for (int kk = 0; kk < 32; ++kk) {
    int k = sub * 32 + kk;
    float v = base_val(k, g0, g1, bwf, pts, bwg);
    float denom = sqrtf(gss[b * M_ + k]) + ((k < 2 * K1_) ? 1e-5f : 0.f);
    buf[kk] = (__bf16)(v * sqn / denom);
  }
  __bf16* dst = basesbf + ((size_t)b * N_ + n) * M_ + sub * 32;
#pragma unroll
  for (int u = 0; u < 8; ++u) *(bf4v*)(dst + u * 4) = *(bf4v*)(buf + u * 4);
}

// ---------------------------------------------------------------------------
// fc0 + LayerNorm -> h (bf16).  grid = B*N blocks of 128 threads.
// ---------------------------------------------------------------------------
__global__ __launch_bounds__(128) void k_fc0_ln(const float* __restrict__ x,
                                                const float* __restrict__ w,
                                                const float* __restrict__ bias,
                                                const float* __restrict__ g,
                                                const float* __restrict__ be,
                                                __bf16* __restrict__ hbf) {
  int idx = blockIdx.x;
  int b = idx >> 13;
  int n = idx & (N_ - 1);
  int c = threadIdx.x;
  const float* xr = x + ((size_t)b * N_ + n) * 3;
  float v = xr[0] * w[c] + xr[1] * w[C_ + c] + xr[2] * w[2 * C_ + c] + bias[c];
  __shared__ float s1[4], s2[4];
  float a = v, q = v * v;
#pragma unroll
  for (int off = 16; off > 0; off >>= 1) {
    a += __shfl_xor(a, off, 32);
    q += __shfl_xor(q, off, 32);
  }
  int wv = c >> 5;
  if ((c & 31) == 0) { s1[wv] = a; s2[wv] = q; }
  __syncthreads();
  float S = s1[0] + s1[1] + s1[2] + s1[3];
  float Q = s2[0] + s2[1] + s2[2] + s2[3];
  float mean = S * (1.f / C_);
  float var = Q * (1.f / C_) - mean * mean;
  float xn = (v - mean) * rsqrtf(var + 1e-5f) * g[c] + be[c];
  hbf[((size_t)b * N_ + n) * C_ + c] = (__bf16)xn;
}

// ---------------------------------------------------------------------------
// Weight repacks
// ---------------------------------------------------------------------------
__global__ void k_pack_spw(const float* __restrict__ sp, __bf16* __restrict__ o) {
  // out[((l*KM+j)*C+o)*C+i] = sp[((l*C+i)*C+o)*KM+j]
  int total = L_ * KM_ * C_ * C_;
  for (int idx = blockIdx.x * blockDim.x + threadIdx.x; idx < total;
       idx += gridDim.x * blockDim.x) {
    int i = idx & (C_ - 1);
    int oc = (idx >> 7) & (C_ - 1);
    int j = (idx >> 14) & (KM_ - 1);
    int l = idx >> 18;
    o[idx] = (__bf16)sp[(((size_t)l * C_ + i) * C_ + oc) * KM_ + j];
  }
}
__global__ void k_pack_h1t(const float* __restrict__ h1, __bf16* __restrict__ o) {
  // out[k*(KM*M) + j*M + l] = h1[(j*M+k)*M+l]   (n-major for GEMM3 B)
  int total = KM_ * M_ * M_;
  for (int idx = blockIdx.x * blockDim.x + threadIdx.x; idx < total;
       idx += gridDim.x * blockDim.x) {
    int l = idx & (M_ - 1);
    int j = (idx >> 8) & (KM_ - 1);
    int k = idx >> 12;
    o[idx] = (__bf16)h1[((size_t)j * M_ + k) * M_ + l];
  }
}
__global__ void k_pack_fc1t(const float* __restrict__ w, __bf16* __restrict__ o) {
  // out[c*C + i] = w[i*C + c]
  int total = C_ * C_;
  for (int idx = blockIdx.x * blockDim.x + threadIdx.x; idx < total;
       idx += gridDim.x * blockDim.x) {
    int i = idx & (C_ - 1);
    int c = idx >> 7;
    o[idx] = (__bf16)w[(size_t)i * C_ + c];
  }
}
__global__ void k_pack_cvt(const float* __restrict__ in, __bf16* __restrict__ o, int total) {
  for (int idx = blockIdx.x * blockDim.x + threadIdx.x; idx < total;
       idx += gridDim.x * blockDim.x)
    o[idx] = (__bf16)in[idx];
}

// ---------------------------------------------------------------------------
// GEMM1: xht[b,l,i] = (1/N) sum_n bases[b,n,l] * h[b,n,i]    (K = N = 8192)
// Output stored transposed [l][i] = exactly the n-major layout GEMM2 needs.
// Both sources are K-major -> transpose during LDS fill: each thread owns a
// K-pair and packs (k,k+1) into dword ds_store_b32.  K-step 64.
// grid = B*4*2 = 64 blocks (64x64 tile each)
// ---------------------------------------------------------------------------
__global__ __launch_bounds__(256) void k_gemm_xh(const __bf16* __restrict__ hbf,
                                                 const __bf16* __restrict__ basesbf,
                                                 __bf16* __restrict__ xht) {
  __shared__ __bf16 ldsA[64 * 72];  // [m = l (64)][k = n (64)]
  __shared__ __bf16 ldsB[64 * 72];  // [n = i (64)][k = n (64)]
  int blk = blockIdx.x;
  int b = blk >> 3;
  int m0 = ((blk >> 1) & 3) * 64;  // over M=256
  int n0 = (blk & 1) * 64;         // over C=128
  int t = threadIdx.x, wave = t >> 5, lane = t & 31, m = lane & 15, half = lane >> 4;
  int wm = wave & 3, wn = (wave >> 2) * 2;
  int kp = t >> 3, ms = t & 7;  // K-pair 0..31, m-segment 0..7
  v8f acc0 = {}, acc1 = {};
  for (int kk0 = 0; kk0 < N_; kk0 += 64) {
    size_t rA = ((size_t)b * N_ + kk0 + 2 * kp) * M_ + m0 + ms * 8;
    size_t rB = ((size_t)b * N_ + kk0 + 2 * kp) * C_ + n0 + ms * 8;
    bf8v a0 = *(const bf8v*)&basesbf[rA];
    bf8v a1 = *(const bf8v*)&basesbf[rA + M_];
    bf8v h0 = *(const bf8v*)&hbf[rB];
    bf8v h1 = *(const bf8v*)&hbf[rB + C_];
    if (kk0 + 192 <= N_) {  // prefetch 2 stages ahead (global_prefetch path)
      __builtin_prefetch(&basesbf[rA + (size_t)128 * M_], 0, 1);
      __builtin_prefetch(&hbf[rB + (size_t)128 * C_], 0, 1);
    }
#pragma unroll
    for (int j2 = 0; j2 < 8; ++j2) {
      bf2v pa = {a0[j2], a1[j2]};
      bf2v ph = {h0[j2], h1[j2]};
      *(bf2v*)&ldsA[(ms * 8 + j2) * 72 + 2 * kp] = pa;
      *(bf2v*)&ldsB[(ms * 8 + j2) * 72 + 2 * kp] = ph;
    }
    __syncthreads();
#pragma unroll
    for (int ks = 0; ks < 64; ks += 32) {
      v16bf a = ld_a_mk2(ldsA + wm * 16 * 72 + ks, 72, m, half);
      v16bf b0 = ld_b_nk2(ldsB + wn * 16 * 72 + ks, 72, m, half);
      v16bf b1 = ld_b_nk2(ldsB + (wn + 1) * 16 * 72 + ks, 72, m, half);
      acc0 = WMMA_BF16(a, b0, acc0);
      acc1 = WMMA_BF16(a, b1, acc1);
    }
    __syncthreads();
  }
  const float alpha = 1.0f / (float)N_;
#pragma unroll
  for (int r = 0; r < 8; ++r) {
    int row = m0 + wm * 16 + half * 8 + r;  // l
    int col = n0 + wn * 16 + (lane & 15);   // i
    xht[((size_t)b * M_ + row) * C_ + col] = (__bf16)(acc0[r] * alpha);
    xht[((size_t)b * M_ + row) * C_ + col + 16] = (__bf16)(acc1[r] * alpha);
  }
}

// ---------------------------------------------------------------------------
// GEMM2: t[b,o,j,l] = sum_i spw[layer][j][o][i] * xht[b,l,i]   (K = 128)
// grid = (B*KM)*8 = 1024 blocks
// ---------------------------------------------------------------------------
__global__ __launch_bounds__(256) void k_gemm_t(const __bf16* __restrict__ spwbf,
                                                const __bf16* __restrict__ xht,
                                                __bf16* __restrict__ tbf, int layer) {
  __shared__ __bf16 ldsA[64 * 40];  // [m = o][k = i]
  __shared__ __bf16 ldsB[64 * 40];  // [n = l][k = i]
  int blk = blockIdx.x;
  int bj = blk >> 3;
  int b = bj >> 4, j = bj & 15;
  int m0 = ((blk >> 2) & 1) * 64;
  int n0 = (blk & 3) * 64;
  int t = threadIdx.x, wave = t >> 5, lane = t & 31, m = lane & 15, half = lane >> 4;
  int wm = wave & 3, wn = (wave >> 2) * 2;
  v8f acc0 = {}, acc1 = {};
  const __bf16* aSrc = spwbf + ((size_t)(layer * KM_ + j) * C_) * C_;
  int mm = t >> 2, ks = t & 3;
  for (int kk0 = 0; kk0 < C_; kk0 += 32) {
    *(bf8v*)&ldsA[mm * 40 + ks * 8] =
        *(const bf8v*)&aSrc[(size_t)(m0 + mm) * C_ + kk0 + ks * 8];
    *(bf8v*)&ldsB[mm * 40 + ks * 8] =
        *(const bf8v*)&xht[((size_t)b * M_ + n0 + mm) * C_ + kk0 + ks * 8];
    __syncthreads();
    v16bf a = ld_a_mk2(ldsA + wm * 16 * 40, 40, m, half);
    v16bf b0 = ld_b_nk2(ldsB + wn * 16 * 40, 40, m, half);
    v16bf b1 = ld_b_nk2(ldsB + (wn + 1) * 16 * 40, 40, m, half);
    acc0 = WMMA_BF16(a, b0, acc0);
    acc1 = WMMA_BF16(a, b1, acc1);
    __syncthreads();
  }
#pragma unroll
  for (int r = 0; r < 8; ++r) {
    int row = m0 + wm * 16 + half * 8 + r;
    int col = n0 + wn * 16 + (lane & 15);
    size_t base = (((size_t)b * C_ + row) * KM_ + j) * M_;
    tbf[base + col] = (__bf16)acc0[r];
    tbf[base + col + 16] = (__bf16)acc1[r];
  }
}

// ---------------------------------------------------------------------------
// GEMM3: y[b,o,k] = sum_{jl} t[b,o,jl] * h1tt[k,jl]   (K = KM*M = 4096)
// grid = B*8 = 64 blocks
// ---------------------------------------------------------------------------
__global__ __launch_bounds__(256) void k_gemm_y(const __bf16* __restrict__ tbf,
                                                const __bf16* __restrict__ h1tt,
                                                __bf16* __restrict__ ybf) {
  __shared__ __bf16 ldsA[64 * 40];  // [m = o][k = jl]
  __shared__ __bf16 ldsB[64 * 40];  // [n = kM][k = jl]
  int blk = blockIdx.x;
  int b = blk >> 3;
  int m0 = ((blk >> 2) & 1) * 64;
  int n0 = (blk & 3) * 64;
  int t = threadIdx.x, wave = t >> 5, lane = t & 31, m = lane & 15, half = lane >> 4;
  int wm = wave & 3, wn = (wave >> 2) * 2;
  v8f acc0 = {}, acc1 = {};
  const int KD = KM_ * M_;
  int mm = t >> 2, ks = t & 3;
  for (int kk0 = 0; kk0 < KD; kk0 += 32) {
    size_t rA = ((size_t)b * C_ + m0 + mm) * KD + kk0 + ks * 8;
    size_t rB = (size_t)(n0 + mm) * KD + kk0 + ks * 8;
    *(bf8v*)&ldsA[mm * 40 + ks * 8] = *(const bf8v*)&tbf[rA];
    *(bf8v*)&ldsB[mm * 40 + ks * 8] = *(const bf8v*)&h1tt[rB];
    if (kk0 + 96 <= KD) {
      __builtin_prefetch(&tbf[rA + 64], 0, 1);
      __builtin_prefetch(&h1tt[rB + 64], 0, 1);
    }
    __syncthreads();
    v16bf a = ld_a_mk2(ldsA + wm * 16 * 40, 40, m, half);
    v16bf b0 = ld_b_nk2(ldsB + wn * 16 * 40, 40, m, half);
    v16bf b1 = ld_b_nk2(ldsB + (wn + 1) * 16 * 40, 40, m, half);
    acc0 = WMMA_BF16(a, b0, acc0);
    acc1 = WMMA_BF16(a, b1, acc1);
    __syncthreads();
  }
#pragma unroll
  for (int r = 0; r < 8; ++r) {
    int row = m0 + wm * 16 + half * 8 + r;
    int col = n0 + wn * 16 + (lane & 15);
    ybf[((size_t)b * C_ + row) * M_ + col] = (__bf16)acc0[r];
    ybf[((size_t)b * C_ + row) * M_ + col + 16] = (__bf16)acc1[r];
  }
}

// ---------------------------------------------------------------------------
// GEMM4 fused: h_new = LN( bases@y^T + h@ws_w^T + ws_b ) (+GELU if layer<L-1)
// Block = 64 rows x full C=128 cols.  K = 256 (bases x y) then 128 (h x ws_w).
// grid = B*128 = 1024 blocks
// ---------------------------------------------------------------------------
__global__ __launch_bounds__(256) void k_gemm_x1h(const __bf16* __restrict__ basesbf,
                                                  const __bf16* __restrict__ ybf,
                                                  const __bf16* __restrict__ hin,
                                                  const __bf16* __restrict__ wswbf,
                                                  const float* __restrict__ wsb,
                                                  const float* __restrict__ lng,
                                                  const float* __restrict__ lnb,
                                                  __bf16* __restrict__ hout, int layer) {
  __shared__ __bf16 ldsA[64 * 40];   // [m = n-row][k]
  __shared__ __bf16 ldsB[128 * 40];  // [n = c][k]
  __shared__ float outT[64 * 132];
  int blk = blockIdx.x;
  int b = blk >> 7;
  int r0 = (blk & 127) * 64;
  int t = threadIdx.x, wave = t >> 5, lane = t & 31, m = lane & 15, half = lane >> 4;
  int wr = wave >> 1, cg = (wave & 1) * 64;
  v8f acc[4] = {{}, {}, {}, {}};
  int mm = t >> 2, ks = t & 3;
  int cB = t >> 1, ksB = t & 1;
  for (int step = 0; step < 12; ++step) {
    int kk = step * 32;
    if (kk < M_) {
      *(bf8v*)&ldsA[mm * 40 + ks * 8] =
          *(const bf8v*)&basesbf[((size_t)b * N_ + r0 + mm) * M_ + kk + ks * 8];
      *(bf8v*)&ldsB[cB * 40 + ksB * 16] =
          *(const bf8v*)&ybf[((size_t)b * C_ + cB) * M_ + kk + ksB * 16];
      *(bf8v*)&ldsB[cB * 40 + ksB * 16 + 8] =
          *(const bf8v*)&ybf[((size_t)b * C_ + cB) * M_ + kk + ksB * 16 + 8];
    } else {
      int k2 = kk - M_;
      *(bf8v*)&ldsA[mm * 40 + ks * 8] =
          *(const bf8v*)&hin[((size_t)b * N_ + r0 + mm) * C_ + k2 + ks * 8];
      *(bf8v*)&ldsB[cB * 40 + ksB * 16] =
          *(const bf8v*)&wswbf[((size_t)layer * C_ + cB) * C_ + k2 + ksB * 16];
      *(bf8v*)&ldsB[cB * 40 + ksB * 16 + 8] =
          *(const bf8v*)&wswbf[((size_t)layer * C_ + cB) * C_ + k2 + ksB * 16 + 8];
    }
    __syncthreads();
    v16bf a = ld_a_mk2(ldsA + wr * 16 * 40, 40, m, half);
#pragma unroll
    for (int jt = 0; jt < 4; ++jt) {
      v16bf bb = ld_b_nk2(ldsB + (cg + jt * 16) * 40, 40, m, half);
      acc[jt] = WMMA_BF16(a, bb, acc[jt]);
    }
    __syncthreads();
  }
#pragma unroll
  for (int jt = 0; jt < 4; ++jt)
#pragma unroll
    for (int r = 0; r < 8; ++r) {
      int rr = wr * 16 + half * 8 + r;
      int cc = cg + jt * 16 + (lane & 15);
      outT[rr * 132 + cc] = acc[jt][r] + wsb[layer * C_ + cc];
    }
  __syncthreads();
  for (int i = 0; i < 8; ++i) {
    int lr = wave * 8 + i;
    float v[4];
    float s = 0.f, sq = 0.f;
#pragma unroll
    for (int u = 0; u < 4; ++u) {
      v[u] = outT[lr * 132 + lane * 4 + u];
      s += v[u];
      sq += v[u] * v[u];
    }
#pragma unroll
    for (int off = 16; off > 0; off >>= 1) {
      s += __shfl_xor(s, off, 32);
      sq += __shfl_xor(sq, off, 32);
    }
    float mean = s * (1.f / C_);
    float var = sq * (1.f / C_) - mean * mean;
    float rstd = rsqrtf(var + 1e-5f);
    __bf16 o4[4];
#pragma unroll
    for (int u = 0; u < 4; ++u) {
      int cc = lane * 4 + u;
      float xn = (v[u] - mean) * rstd * lng[layer * C_ + cc] + lnb[layer * C_ + cc];
      if (layer < L_ - 1) xn = 0.5f * xn * (1.f + erff(xn * 0.70710678118f));
      o4[u] = (__bf16)xn;
    }
    *(bf4v*)&hout[((size_t)b * N_ + r0 + lr) * C_ + lane * 4] = *(bf4v*)o4;
  }
}

// ---------------------------------------------------------------------------
// GEMM5 fused: out[b,n] = gelu(h@fc1_w + fc1_b) . fc2_w + fc2_b  (K = 128)
// fc1 pre-transposed to [c][i].  grid = B*128 = 1024 blocks
// ---------------------------------------------------------------------------
__global__ __launch_bounds__(256) void k_gemm_fc12(const __bf16* __restrict__ hbf,
                                                   const __bf16* __restrict__ fc1t,
                                                   const float* __restrict__ fc1b,
                                                   const float* __restrict__ fc2w,
                                                   const float* __restrict__ fc2b,
                                                   float* __restrict__ out) {
  __shared__ __bf16 ldsA[64 * 40];   // [m = n-row][k = i]
  __shared__ __bf16 ldsB[128 * 40];  // [n = c][k = i]
  __shared__ float outT[64 * 132];
  int blk = blockIdx.x;
  int b = blk >> 7;
  int r0 = (blk & 127) * 64;
  int t = threadIdx.x, wave = t >> 5, lane = t & 31, m = lane & 15, half = lane >> 4;
  int wr = wave >> 1, cg = (wave & 1) * 64;
  v8f acc[4] = {{}, {}, {}, {}};
  int mm = t >> 2, ks = t & 3;
  int cB = t >> 1, ksB = t & 1;
  for (int kk = 0; kk < C_; kk += 32) {
    *(bf8v*)&ldsA[mm * 40 + ks * 8] =
        *(const bf8v*)&hbf[((size_t)b * N_ + r0 + mm) * C_ + kk + ks * 8];
    *(bf8v*)&ldsB[cB * 40 + ksB * 16] =
        *(const bf8v*)&fc1t[(size_t)cB * C_ + kk + ksB * 16];
    *(bf8v*)&ldsB[cB * 40 + ksB * 16 + 8] =
        *(const bf8v*)&fc1t[(size_t)cB * C_ + kk + ksB * 16 + 8];
    __syncthreads();
    v16bf a = ld_a_mk2(ldsA + wr * 16 * 40, 40, m, half);
#pragma unroll
    for (int jt = 0; jt < 4; ++jt) {
      v16bf bb = ld_b_nk2(ldsB + (cg + jt * 16) * 40, 40, m, half);
      acc[jt] = WMMA_BF16(a, bb, acc[jt]);
    }
    __syncthreads();
  }
#pragma unroll
  for (int jt = 0; jt < 4; ++jt)
#pragma unroll
    for (int r = 0; r < 8; ++r) {
      int rr = wr * 16 + half * 8 + r;
      int cc = cg + jt * 16 + (lane & 15);
      outT[rr * 132 + cc] = acc[jt][r] + fc1b[cc];
    }
  __syncthreads();
  for (int i = 0; i < 8; ++i) {
    int lr = wave * 8 + i;
    float d = 0.f;
#pragma unroll
    for (int u = 0; u < 4; ++u) {
      int cc = lane * 4 + u;
      float g = outT[lr * 132 + cc];
      g = 0.5f * g * (1.f + erff(g * 0.70710678118f));
      d += g * fc2w[cc];
    }
#pragma unroll
    for (int off = 16; off > 0; off >>= 1) d += __shfl_xor(d, off, 32);
    if (lane == 0) out[(size_t)b * N_ + r0 + lr] = d + fc2b[0];
  }
}

// ---------------------------------------------------------------------------
// Host launcher
// ---------------------------------------------------------------------------
extern "C" void kernel_launch(void* const* d_in, const int* in_sizes, int n_in,
                              void* d_out, int out_size, void* d_ws, size_t ws_size,
                              hipStream_t stream) {
  (void)in_sizes; (void)n_in; (void)out_size; (void)ws_size;
  const float* x    = (const float*)d_in[0];
  const float* bwf  = (const float*)d_in[1];
  const float* pts  = (const float*)d_in[2];
  const float* bwg  = (const float*)d_in[3];
  const float* fc0w = (const float*)d_in[4];
  const float* fc0b = (const float*)d_in[5];
  const float* ln0g = (const float*)d_in[6];
  const float* ln0b = (const float*)d_in[7];
  const float* H1   = (const float*)d_in[8];
  const float* spw  = (const float*)d_in[9];
  const float* wsw  = (const float*)d_in[10];
  const float* wsb  = (const float*)d_in[11];
  const float* lng  = (const float*)d_in[12];
  const float* lnb  = (const float*)d_in[13];
  const float* fc1w = (const float*)d_in[14];
  const float* fc1b = (const float*)d_in[15];
  const float* fc2w = (const float*)d_in[16];
  const float* fc2b = (const float*)d_in[17];

  char* p = (char*)d_ws;
  auto take = [&](size_t bytes) -> char* {
    char* r = p;
    p += (bytes + 255) & ~(size_t)255;
    return r;
  };
  float*  sumsq   = (float*)take((size_t)B_ * M_ * 4);
  __bf16* basesbf = (__bf16*)take((size_t)B_ * N_ * M_ * 2);
  __bf16* hbf0    = (__bf16*)take((size_t)B_ * N_ * C_ * 2);
  __bf16* hbf1    = (__bf16*)take((size_t)B_ * N_ * C_ * 2);
  __bf16* xht     = (__bf16*)take((size_t)B_ * M_ * C_ * 2);
  __bf16* tbf     = (__bf16*)take((size_t)B_ * C_ * KM_ * M_ * 2);
  __bf16* ybf     = (__bf16*)take((size_t)B_ * C_ * M_ * 2);
  __bf16* spwbf   = (__bf16*)take((size_t)L_ * KM_ * C_ * C_ * 2);
  __bf16* h1tt    = (__bf16*)take((size_t)KM_ * M_ * M_ * 2);
  __bf16* wswbf   = (__bf16*)take((size_t)L_ * C_ * C_ * 2);
  __bf16* fc1t    = (__bf16*)take((size_t)C_ * C_ * 2);
  __bf16* hbuf[2] = {hbf0, hbf1};

  hipMemsetAsync(sumsq, 0, (size_t)B_ * M_ * 4, stream);
  k_bases_ss<<<dim3(2048), dim3(256), 0, stream>>>(x, bwf, pts, bwg, sumsq);
  k_bases_write<<<dim3(2048), dim3(256), 0, stream>>>(x, bwf, pts, bwg, sumsq, basesbf);
  k_fc0_ln<<<dim3(B_ * N_), dim3(128), 0, stream>>>(x, fc0w, fc0b, ln0g, ln0b, hbf0);
  k_pack_spw<<<dim3(512), dim3(256), 0, stream>>>(spw, spwbf);
  k_pack_h1t<<<dim3(512), dim3(256), 0, stream>>>(H1, h1tt);
  k_pack_cvt<<<dim3(64), dim3(256), 0, stream>>>(wsw, wswbf, L_ * C_ * C_);
  k_pack_fc1t<<<dim3(32), dim3(256), 0, stream>>>(fc1w, fc1t);

  int cur = 0;
  for (int layer = 0; layer < L_; ++layer) {
    k_gemm_xh<<<dim3(64), dim3(256), 0, stream>>>(hbuf[cur], basesbf, xht);
    k_gemm_t<<<dim3(1024), dim3(256), 0, stream>>>(spwbf, xht, tbf, layer);
    k_gemm_y<<<dim3(64), dim3(256), 0, stream>>>(tbf, h1tt, ybf);
    k_gemm_x1h<<<dim3(1024), dim3(256), 0, stream>>>(basesbf, ybf, hbuf[cur], wswbf, wsb,
                                                     lng, lnb, hbuf[1 - cur], layer);
    cur ^= 1;
  }
  k_gemm_fc12<<<dim3(1024), dim3(256), 0, stream>>>(hbuf[cur], fc1t, fc1b, fc2w, fc2b,
                                                    (float*)d_out);
}
